// Interaction_MLP4d_layer_9698036155141
// MI455X (gfx1250) — matrix operand order repaired and checked by reading, not compiled, not run on hardware
//
#include <hip/hip_runtime.h>
#include <hip/hip_bf16.h>

typedef _Float16 f16;
typedef __attribute__((ext_vector_type(16))) _Float16 v16h;
typedef __attribute__((ext_vector_type(8)))  float    v8f;
typedef __attribute__((ext_vector_type(4)))  unsigned v4u;
typedef __attribute__((ext_vector_type(4)))  int      v4i;
typedef __attribute__((ext_vector_type(8)))  int      v8i;
typedef __attribute__((address_space(3)))    f16      lds_f16_t;

#if defined(__has_builtin)
#if __has_builtin(__builtin_amdgcn_tensor_load_to_lds) && \
    __has_builtin(__builtin_amdgcn_s_wait_tensorcnt)
#define HAVE_TDM 1
#endif
#endif

// ---------------- workspace layout (bytes, all 256-aligned) ----------------
#define OFF_NFT      0u          // f16 [B][N][F][C] = 2*128*32*32 halves = 524288 B
#define OFF_WE_CONV  524288u     // f16 [32][64]
#define OFF_WE_FC1   528384u     // f16 [32][32]
#define OFF_WE_FC2   530432u
#define OFF_WN_CONV  532480u
#define OFF_WN_FC1   534528u
#define OFF_WN_FC2   536576u
#define OFF_WNE_CONV 538624u     // f16 [32][64]
#define OFF_WNE_FC1  542720u
#define OFF_WNE_FC2  544768u
#define OFF_PSUM     548864u     // f32 [B][128][32]
#define OFF_PSQ      581632u     // f32 [B][128][32]
#define OFF_MU_E     614400u     // f32 [B*32]
#define OFF_RS_E     614656u     // f32 [B*32]
#define OFF_INC      614912u     // f32 [B][32][128][32] = 1 MB
#define OFF_X3N      1663488u    // f32 [B][32][128][32]
#define OFF_X3NE     2712064u    // f32 [B][32][128][32]   (total ~3.6 MB)

#define NE_EDGES 16256
#define LN_EPS   1e-5f

__device__ __forceinline__ v8f vz8() {
  v8f z;
#pragma unroll
  for (int i = 0; i < 8; ++i) z[i] = 0.f;
  return z;
}

__device__ __forceinline__ v8f wmma16(v16h a, v16h b, v8f c) {
  return __builtin_amdgcn_wmma_f32_16x16x32_f16(false, a, false, b, (short)0, c,
                                                false, false);
}

__device__ __forceinline__ float siluf(float x) { return x / (1.f + __expf(-x)); }
__device__ __forceinline__ float sigm(float x)  { return 1.f / (1.f + __expf(-x)); }

// A fragment (16x32 f16): lane -> row m = mtile*16 + (lane&15);
// comps i -> k = kbase + 16*(lane>>4) + i (contiguous).
__device__ __forceinline__ v16h load_afrag(const f16* W, int ld, int mtile,
                                           int kbase, int lane) {
  const f16* p = W + (mtile * 16 + (lane & 15)) * ld + kbase + ((lane >> 4) << 4);
  v16h a;
#pragma unroll
  for (int i = 0; i < 16; ++i) a[i] = p[i];
  return a;
}

// B fragment (32x16 f16): source laid out S[n][k] with ld==32 (so B[k][n]=S[n*32+k]).
// lane -> col n = ntile*16 + (lane&15); comps i -> k = 16*(lane>>4) + i (contiguous 32B).
__device__ __forceinline__ v16h load_bfrag(const f16* S, int ntile, int lane) {
  const f16* p = S + (ntile * 16 + (lane & 15)) * 32 + ((lane >> 4) << 4);
  return *(const v16h*)p;
}

// conv-D (+bias) -> fc1(+bias,silu) -> fc2(+bias,silu) ; biasL = [cb(32)|f1b(32)|f2b(32)]
__device__ __forceinline__ void mlp_tail(v8f dc[2][2], const float* biasL,
                                         const v16h b1[2], const v16h b2[2],
                                         f16* xbuf, int lane, v8f x3[2][2]) {
  const int r0 = (lane >> 4) * 8, nn = lane & 15;
#pragma unroll
  for (int ht = 0; ht < 2; ++ht)
#pragma unroll
    for (int ft = 0; ft < 2; ++ft)
#pragma unroll
      for (int r = 0; r < 8; ++r)
        xbuf[(ht * 16 + r0 + r) * 32 + ft * 16 + nn] =
            (f16)(dc[ht][ft][r] + biasL[ht * 16 + r0 + r]);
  __syncthreads();
  v16h a0 = load_afrag(xbuf, 32, 0, 0, lane);
  v16h a1 = load_afrag(xbuf, 32, 1, 0, lane);
  v8f d00 = vz8(), d01 = vz8(), d10 = vz8(), d11 = vz8();
  d00 = wmma16(a0, b1[0], d00); d01 = wmma16(a0, b1[1], d01);
  d10 = wmma16(a1, b1[0], d10); d11 = wmma16(a1, b1[1], d11);
  __syncthreads();
#pragma unroll
  for (int r = 0; r < 8; ++r) {
    xbuf[(r0 + r) * 32 + nn]           = (f16)siluf(d00[r] + biasL[32 + nn]);
    xbuf[(r0 + r) * 32 + 16 + nn]      = (f16)siluf(d01[r] + biasL[48 + nn]);
    xbuf[(16 + r0 + r) * 32 + nn]      = (f16)siluf(d10[r] + biasL[32 + nn]);
    xbuf[(16 + r0 + r) * 32 + 16 + nn] = (f16)siluf(d11[r] + biasL[48 + nn]);
  }
  __syncthreads();
  a0 = load_afrag(xbuf, 32, 0, 0, lane);
  a1 = load_afrag(xbuf, 32, 1, 0, lane);
  v8f e00 = vz8(), e01 = vz8(), e10 = vz8(), e11 = vz8();
  e00 = wmma16(a0, b2[0], e00); e01 = wmma16(a0, b2[1], e01);
  e10 = wmma16(a1, b2[0], e10); e11 = wmma16(a1, b2[1], e11);
#pragma unroll
  for (int r = 0; r < 8; ++r) {
    x3[0][0][r] = siluf(e00[r] + biasL[64 + nn]);
    x3[0][1][r] = siluf(e01[r] + biasL[80 + nn]);
    x3[1][0][r] = siluf(e10[r] + biasL[64 + nn]);
    x3[1][1][r] = siluf(e11[r] + biasL[80 + nn]);
  }
  __syncthreads();
}

// ---------------- prep: transpose node_feature to f16 [b][n][f][c]; weights to f16 ----
__global__ __launch_bounds__(256) void prep_kernel(
    const float* nf, const float* wec, const float* we1, const float* we2,
    const float* wnc, const float* wn1, const float* wn2, const float* wnec,
    const float* wne1, const float* wne2, char* ws) {
  int idx = blockIdx.x * 256 + threadIdx.x;
  f16* nft = (f16*)(ws + OFF_NFT);
  if (idx < 262144) {
    int c = idx & 31, f = (idx >> 5) & 31, n = (idx >> 10) & 127, b = idx >> 17;
    nft[idx] = (f16)nf[(((size_t)b * 32 + c) * 128 + n) * 32 + f];
  }
  if (idx < 2048) {
    ((f16*)(ws + OFF_WE_CONV))[idx]  = (f16)wec[idx];
    ((f16*)(ws + OFF_WNE_CONV))[idx] = (f16)wnec[idx];
  }
  if (idx < 1024) {
    ((f16*)(ws + OFF_WE_FC1))[idx]  = (f16)we1[idx];
    ((f16*)(ws + OFF_WE_FC2))[idx]  = (f16)we2[idx];
    ((f16*)(ws + OFF_WN_CONV))[idx] = (f16)wnc[idx];
    ((f16*)(ws + OFF_WN_FC1))[idx]  = (f16)wn1[idx];
    ((f16*)(ws + OFF_WN_FC2))[idx]  = (f16)wn2[idx];
    ((f16*)(ws + OFF_WNE_FC1))[idx] = (f16)wne1[idx];
    ((f16*)(ws + OFF_WNE_FC2))[idx] = (f16)wne2[idx];
  }
}

// ---------------- edge kernel: MODE 0 = LN stats pass, MODE 1 = output pass ----------
template <int MODE>
__global__ __launch_bounds__(256) void edge_kernel(
    const float* ecb, const float* e1b, const float* e2b, const float* edges_mask,
    const float* elng, const float* elnb, const float* attw, const float* attb,
    float* d_out, char* ws) {
  const int rec = blockIdx.x & 127;
  const int b   = blockIdx.x >> 7;
  const int tid = threadIdx.x, wave = tid >> 5, lane = tid & 31;
  const int r0 = (lane >> 4) * 8, nn = lane & 15;

  __shared__ alignas(32) f16 xbufS[8][1024];
  __shared__ alignas(32) f16 recvL[1024];   // receiver row [f][c], staged once per block
  __shared__ float biasL[96];
  __shared__ float redS[32], redQ[32];
  __shared__ float muL[32], rsL[32], awL[32];
  __shared__ float accIn[1024];
  __shared__ float sAttB;

  const f16* nf_b = (const f16*)(ws + OFF_NFT) + (size_t)b * 131072;

  // ---- stage receiver row (2 KB) into LDS via the Tensor Data Mover -----------------
#ifdef HAVE_TDM
  if (tid < 32) {  // one TDM op issued by wave 0 (EXEC ignored by tensor ops)
    unsigned long long ga =
        (unsigned long long)(const void*)(nf_b + ((size_t)rec << 10));
    unsigned ldsoff = (unsigned)(unsigned long long)(lds_f16_t*)&recvL[0];
    v4u g0;
    g0[0] = 1u;                                     // count=1 (valid descriptor)
    g0[1] = ldsoff;                                 // lds_addr
    g0[2] = (unsigned)ga;                           // global_addr[31:0]
    g0[3] = ((unsigned)(ga >> 32) & 0x01FFFFFFu) | 0x80000000u;  // addr[56:32]|type=2
    v8i g1;
    g1[0] = (int)(3u << 16);     // workgroup_mask=0, data_size=3 (8B)
    g1[1] = (int)(256u << 16);   // tensor_dim0 = 256 elements (2 KB)
    g1[2] = (int)(1u << 16);     // tensor_dim1 = 1
    g1[3] = (int)(256u << 16);   // tile_dim0 = 256
    g1[4] = 0;                   // tile_dim1/2 unused (1-D tile)
    g1[5] = 256;                 // tensor_dim0_stride
    g1[6] = 0;
    g1[7] = 0;
    v4i g2 = {0, 0, 0, 0};
    v4i g3 = {0, 0, 0, 0};
#if __clang_major__ >= 23
    v8i g4 = {0, 0, 0, 0, 0, 0, 0, 0};
    __builtin_amdgcn_tensor_load_to_lds(g0, g1, g2, g3, g4, 0);
#else
    __builtin_amdgcn_tensor_load_to_lds(g0, g1, g2, g3, 0);
#endif
    __builtin_amdgcn_s_wait_tensorcnt(0);
  }
#else
  {
    const f16* src = nf_b + ((size_t)rec << 10);
    for (int j = tid; j < 1024; j += 256) recvL[j] = src[j];
  }
#endif

  if (tid < 32) {
    biasL[tid] = ecb[tid]; biasL[32 + tid] = e1b[tid]; biasL[64 + tid] = e2b[tid];
    redS[tid] = 0.f; redQ[tid] = 0.f;
    if (MODE == 1) {
      muL[tid] = ((const float*)(ws + OFF_MU_E))[b * 32 + tid];
      rsL[tid] = ((const float*)(ws + OFF_RS_E))[b * 32 + tid];
      awL[tid] = attw[tid];
    }
  }
  if (MODE == 1) {
    for (int j = tid; j < 1024; j += 256) accIn[j] = 0.f;
    if (tid == 0) sAttB = attb[0];
  }

  const f16* wc = (const f16*)(ws + OFF_WE_CONV);
  const f16* w1 = (const f16*)(ws + OFF_WE_FC1);
  const f16* w2 = (const f16*)(ws + OFF_WE_FC2);
  v16h ac[2][2];
#pragma unroll
  for (int ht = 0; ht < 2; ++ht)
#pragma unroll
    for (int ks = 0; ks < 2; ++ks) ac[ht][ks] = load_afrag(wc, 64, ht, ks * 32, lane);
  v16h b1[2] = {load_bfrag(w1, 0, lane), load_bfrag(w1, 1, lane)};
  v16h b2[2] = {load_bfrag(w2, 0, lane), load_bfrag(w2, 1, lane)};
  __syncthreads();

  // receiver B fragments are loop-invariant: hoist out of the edge loop
  v16h bR0 = load_bfrag(recvL, 0, lane), bR1 = load_bfrag(recvL, 1, lane);

  float accS[2][8], accQ[2][8];
#pragma unroll
  for (int ht = 0; ht < 2; ++ht)
#pragma unroll
    for (int r = 0; r < 8; ++r) { accS[ht][r] = 0.f; accQ[ht][r] = 0.f; }
  v8f ain[2][2];
  ain[0][0] = vz8(); ain[0][1] = vz8(); ain[1][0] = vz8(); ain[1][1] = vz8();

  for (int it = 0; it < 16; ++it) {
    int el = it * 8 + wave;
    bool valid = el < 127;
    int send = valid ? (el + (el >= rec ? 1 : 0)) : rec;
    {  // prefetch next sender row into caches
      int el2 = el + 8;
      int send2 = (el2 < 127) ? (el2 + (el2 >= rec ? 1 : 0)) : rec;
      __builtin_prefetch((const void*)(nf_b + ((size_t)send2 << 10)), 0, 3);
    }
    v8f dc[2][2];
    dc[0][0] = vz8(); dc[0][1] = vz8(); dc[1][0] = vz8(); dc[1][1] = vz8();
    {
      const f16* s0 = nf_b + ((size_t)send << 10);           // sender: channels 0..31
      v16h bf0 = load_bfrag(s0, 0, lane), bf1 = load_bfrag(s0, 1, lane);
      dc[0][0] = wmma16(ac[0][0], bf0, dc[0][0]);
      dc[0][1] = wmma16(ac[0][0], bf1, dc[0][1]);
      dc[1][0] = wmma16(ac[1][0], bf0, dc[1][0]);
      dc[1][1] = wmma16(ac[1][0], bf1, dc[1][1]);
      dc[0][0] = wmma16(ac[0][1], bR0, dc[0][0]);            // receiver: channels 32..63
      dc[0][1] = wmma16(ac[0][1], bR1, dc[0][1]);
      dc[1][0] = wmma16(ac[1][1], bR0, dc[1][0]);
      dc[1][1] = wmma16(ac[1][1], bR1, dc[1][1]);
    }
    v8f x3[2][2];
    mlp_tail(dc, biasL, b1, b2, xbufS[wave], lane, x3);

    if (MODE == 0) {
      if (valid) {
#pragma unroll
        for (int ht = 0; ht < 2; ++ht)
#pragma unroll
          for (int r = 0; r < 8; ++r) {
            float a = x3[ht][0][r], c = x3[ht][1][r];
            accS[ht][r] += a + c;
            accQ[ht][r] += a * a + c * c;
          }
      }
    } else {
      if (valid) {   // uniform per wave
        int e = rec * 127 + el;
        float em = edges_mask[b * NE_EDGES + e];
        float gg0 = elng[e * 32 + nn],      gg1 = elng[e * 32 + 16 + nn];
        float bb0 = elnb[e * 32 + nn],      bb1 = elnb[e * 32 + 16 + nn];
        float pa0 = 0.f, pa1 = 0.f;
        float* mij = d_out + 262144;
#pragma unroll
        for (int ht = 0; ht < 2; ++ht)
#pragma unroll
          for (int r = 0; r < 8; ++r) {
            int h = ht * 16 + r0 + r;
            float nrm = rsL[h], mu = muL[h], aw = awL[h];
            float m10 = ((x3[ht][0][r] - mu) * nrm * gg0 + bb0) * em;
            float m11 = ((x3[ht][1][r] - mu) * nrm * gg1 + bb1) * em;
            size_t base = ((size_t)(b * 32 + h) * NE_EDGES + e) * 32;
            mij[base + nn]      = m10 * em;
            mij[base + 16 + nn] = m11 * em;
            pa0 += m10 * aw; pa1 += m11 * aw;
            x3[ht][0][r] = m10; x3[ht][1][r] = m11;
          }
        float at0 = sigm(pa0 + __shfl_xor(pa0, 16, 32) + sAttB) * em;
        float at1 = sigm(pa1 + __shfl_xor(pa1, 16, 32) + sAttB) * em;
#pragma unroll
        for (int ht = 0; ht < 2; ++ht)
#pragma unroll
          for (int r = 0; r < 8; ++r) {
            ain[ht][0][r] += x3[ht][0][r] * at0;
            ain[ht][1][r] += x3[ht][1][r] * at1;
          }
      }
    }
  }

  if (MODE == 0) {
#pragma unroll
    for (int ht = 0; ht < 2; ++ht)
#pragma unroll
      for (int r = 0; r < 8; ++r) {
        atomicAdd(&redS[ht * 16 + r0 + r], accS[ht][r]);
        atomicAdd(&redQ[ht * 16 + r0 + r], accQ[ht][r]);
      }
    __syncthreads();
    if (tid < 32) {
      ((float*)(ws + OFF_PSUM))[(b * 128 + rec) * 32 + tid] = redS[tid];
      ((float*)(ws + OFF_PSQ))[(b * 128 + rec) * 32 + tid]  = redQ[tid];
    }
  } else {
#pragma unroll
    for (int ht = 0; ht < 2; ++ht)
#pragma unroll
      for (int ft = 0; ft < 2; ++ft)
#pragma unroll
        for (int r = 0; r < 8; ++r)
          atomicAdd(&accIn[(ht * 16 + r0 + r) * 32 + ft * 16 + nn], ain[ht][ft][r]);
    __syncthreads();
    float* inc = (float*)(ws + OFF_INC);
    for (int j = tid; j < 1024; j += 256) {
      int h = j >> 5, f = j & 31;
      inc[(((size_t)b * 32 + h) * 128 + rec) * 32 + f] = accIn[j];
    }
  }
}

// ---------------- reduce partials -> mu, rsig per (b,h) --------------------------------
__global__ void stats_kernel(char* ws) {
  int t = threadIdx.x;
  if (t < 64) {
    const float* ps = (const float*)(ws + OFF_PSUM);
    const float* pq = (const float*)(ws + OFF_PSQ);
    int b = t >> 5, h = t & 31;
    float s = 0.f, q = 0.f;
    for (int r = 0; r < 128; ++r) {
      s += ps[(b * 128 + r) * 32 + h];
      q += pq[(b * 128 + r) * 32 + h];
    }
    const float inv = 1.f / (float)(NE_EDGES * 32);
    float mu = s * inv;
    float var = q * inv - mu * mu;
    ((float*)(ws + OFF_MU_E))[t] = mu;
    ((float*)(ws + OFF_RS_E))[t] = rsqrtf(var + LN_EPS);
  }
}

// ---------------- node kernel: n-MLP -> LN -> ne-MLP -> LN -> masked out_node ----------
__global__ __launch_bounds__(128) void node_kernel(
    const float* nodes_mask, const float* ncb, const float* n1b, const float* n2b,
    const float* nlng, const float* nlnb, const float* necb, const float* ne1b,
    const float* ne2b, const float* nelng, const float* nelnb, float* d_out, char* ws) {
  const int b = blockIdx.x;
  const int tid = threadIdx.x, wave = tid >> 5, lane = tid & 31;
  const int r0 = (lane >> 4) * 8, nn = lane & 15;
  __shared__ alignas(32) f16 stS[4][1024];
  __shared__ alignas(32) f16 xbufS[4][1024];
  __shared__ float biasL[96];
  __shared__ float redS[32], redQ[32], muL[32], rsL[32];
  const f16* nf_b = (const f16*)(ws + OFF_NFT) + (size_t)b * 131072;
  const float* incoming = (const float*)(ws + OFF_INC);
  float* x3n = (float*)(ws + OFF_X3N);
  float* x3ne = (float*)(ws + OFF_X3NE);

  // ---------------- stage 1: n-MLP ----------------
  if (tid < 32) {
    biasL[tid] = ncb[tid]; biasL[32 + tid] = n1b[tid]; biasL[64 + tid] = n2b[tid];
    redS[tid] = 0.f; redQ[tid] = 0.f;
  }
  const f16* wnc = (const f16*)(ws + OFF_WN_CONV);
  const f16* wn1 = (const f16*)(ws + OFF_WN_FC1);
  const f16* wn2 = (const f16*)(ws + OFF_WN_FC2);
  v16h acn[2] = {load_afrag(wnc, 32, 0, 0, lane), load_afrag(wnc, 32, 1, 0, lane)};
  v16h b1n[2] = {load_bfrag(wn1, 0, lane), load_bfrag(wn1, 1, lane)};
  v16h b2n[2] = {load_bfrag(wn2, 0, lane), load_bfrag(wn2, 1, lane)};
  __syncthreads();

  float accS[2][8], accQ[2][8];
#pragma unroll
  for (int ht = 0; ht < 2; ++ht)
#pragma unroll
    for (int r = 0; r < 8; ++r) { accS[ht][r] = 0.f; accQ[ht][r] = 0.f; }

  for (int it = 0; it < 32; ++it) {
    const int n = it * 4 + wave;
#pragma unroll 4
    for (int h = 0; h < 32; ++h)
      stS[wave][lane * 32 + h] =
          (f16)(incoming[(((size_t)b * 32 + h) * 128 + n) * 32 + lane] * (1.f / 128.f));
    __syncthreads();
    v8f dc[2][2];
    dc[0][0] = vz8(); dc[0][1] = vz8(); dc[1][0] = vz8(); dc[1][1] = vz8();
    v16h bf0 = load_bfrag(stS[wave], 0, lane), bf1 = load_bfrag(stS[wave], 1, lane);
    dc[0][0] = wmma16(acn[0], bf0, dc[0][0]); dc[0][1] = wmma16(acn[0], bf1, dc[0][1]);
    dc[1][0] = wmma16(acn[1], bf0, dc[1][0]); dc[1][1] = wmma16(acn[1], bf1, dc[1][1]);
    v8f x3[2][2];
    mlp_tail(dc, biasL, b1n, b2n, xbufS[wave], lane, x3);
#pragma unroll
    for (int ht = 0; ht < 2; ++ht)
#pragma unroll
      for (int ft = 0; ft < 2; ++ft)
#pragma unroll
        for (int r = 0; r < 8; ++r) {
          int h = ht * 16 + r0 + r, f = ft * 16 + nn;
          x3n[(((size_t)b * 32 + h) * 128 + n) * 32 + f] = x3[ht][ft][r];
        }
#pragma unroll
    for (int ht = 0; ht < 2; ++ht)
#pragma unroll
      for (int r = 0; r < 8; ++r) {
        float a = x3[ht][0][r], c = x3[ht][1][r];
        accS[ht][r] += a + c; accQ[ht][r] += a * a + c * c;
      }
  }
#pragma unroll
  for (int ht = 0; ht < 2; ++ht)
#pragma unroll
    for (int r = 0; r < 8; ++r) {
      atomicAdd(&redS[ht * 16 + r0 + r], accS[ht][r]);
      atomicAdd(&redQ[ht * 16 + r0 + r], accQ[ht][r]);
    }
  __syncthreads();
  if (tid < 32) {
    float mu = redS[tid] * (1.f / 4096.f);
    float var = redQ[tid] * (1.f / 4096.f) - mu * mu;
    muL[tid] = mu; rsL[tid] = rsqrtf(var + LN_EPS);
  }
  __syncthreads();
  if (tid < 32) {
    biasL[tid] = necb[tid]; biasL[32 + tid] = ne1b[tid]; biasL[64 + tid] = ne2b[tid];
    redS[tid] = 0.f; redQ[tid] = 0.f;
  }
  const f16* wnec = (const f16*)(ws + OFF_WNE_CONV);
  const f16* wne1 = (const f16*)(ws + OFF_WNE_FC1);
  const f16* wne2 = (const f16*)(ws + OFF_WNE_FC2);
  v16h acne[2][2];
#pragma unroll
  for (int ht = 0; ht < 2; ++ht)
#pragma unroll
    for (int ks = 0; ks < 2; ++ks) acne[ht][ks] = load_afrag(wnec, 64, ht, ks * 32, lane);
  v16h b1e[2] = {load_bfrag(wne1, 0, lane), load_bfrag(wne1, 1, lane)};
  v16h b2e[2] = {load_bfrag(wne2, 0, lane), load_bfrag(wne2, 1, lane)};
#pragma unroll
  for (int ht = 0; ht < 2; ++ht)
#pragma unroll
    for (int r = 0; r < 8; ++r) { accS[ht][r] = 0.f; accQ[ht][r] = 0.f; }
  __syncthreads();

  // ---------------- stage 2: ne-MLP (concat node_feature + normalized incoming) -------
  for (int it = 0; it < 32; ++it) {
    const int n = it * 4 + wave;
#pragma unroll 4
    for (int h = 0; h < 32; ++h) {
      float x = x3n[(((size_t)b * 32 + h) * 128 + n) * 32 + lane];
      float xi = (x - muL[h]) * rsL[h] * nlng[n * 32 + lane] + nlnb[n * 32 + lane];
      stS[wave][lane * 32 + h] = (f16)xi;
    }
    __syncthreads();
    v8f dc[2][2];
    dc[0][0] = vz8(); dc[0][1] = vz8(); dc[1][0] = vz8(); dc[1][1] = vz8();
    {
      const f16* s0 = nf_b + ((size_t)n << 10);  // channels 0..31: node_feature
      v16h g0 = load_bfrag(s0, 0, lane), g1 = load_bfrag(s0, 1, lane);
      dc[0][0] = wmma16(acne[0][0], g0, dc[0][0]);
      dc[0][1] = wmma16(acne[0][0], g1, dc[0][1]);
      dc[1][0] = wmma16(acne[1][0], g0, dc[1][0]);
      dc[1][1] = wmma16(acne[1][0], g1, dc[1][1]);
      v16h l0 = load_bfrag(stS[wave], 0, lane), l1 = load_bfrag(stS[wave], 1, lane);
      dc[0][0] = wmma16(acne[0][1], l0, dc[0][0]);
      dc[0][1] = wmma16(acne[0][1], l1, dc[0][1]);
      dc[1][0] = wmma16(acne[1][1], l0, dc[1][0]);
      dc[1][1] = wmma16(acne[1][1], l1, dc[1][1]);
    }
    v8f x3[2][2];
    mlp_tail(dc, biasL, b1e, b2e, xbufS[wave], lane, x3);
#pragma unroll
    for (int ht = 0; ht < 2; ++ht)
#pragma unroll
      for (int ft = 0; ft < 2; ++ft)
#pragma unroll
        for (int r = 0; r < 8; ++r) {
          int h = ht * 16 + r0 + r, f = ft * 16 + nn;
          x3ne[(((size_t)b * 32 + h) * 128 + n) * 32 + f] = x3[ht][ft][r];
        }
#pragma unroll
    for (int ht = 0; ht < 2; ++ht)
#pragma unroll
      for (int r = 0; r < 8; ++r) {
        float a = x3[ht][0][r], c = x3[ht][1][r];
        accS[ht][r] += a + c; accQ[ht][r] += a * a + c * c;
      }
  }
#pragma unroll
  for (int ht = 0; ht < 2; ++ht)
#pragma unroll
    for (int r = 0; r < 8; ++r) {
      atomicAdd(&redS[ht * 16 + r0 + r], accS[ht][r]);
      atomicAdd(&redQ[ht * 16 + r0 + r], accQ[ht][r]);
    }
  __syncthreads();
  if (tid < 32) {
    float mu = redS[tid] * (1.f / 4096.f);
    float var = redQ[tid] * (1.f / 4096.f) - mu * mu;
    muL[tid] = mu; rsL[tid] = rsqrtf(var + LN_EPS);
  }
  __syncthreads();

  // ---------------- stage 3: LN + node mask -> out_node --------------------------------
  for (int idx = tid; idx < 131072; idx += 128) {
    int h = idx >> 12, n = (idx >> 5) & 127, f = idx & 31;
    float x = x3ne[(size_t)b * 131072 + idx];
    float v = (x - muL[h]) * rsL[h] * nelng[n * 32 + f] + nelnb[n * 32 + f];
    d_out[(size_t)b * 131072 + idx] = v * nodes_mask[b * 128 + n];
  }
}

// ---------------------------------------------------------------------------------------
extern "C" void kernel_launch(void* const* d_in, const int* in_sizes, int n_in,
                              void* d_out, int out_size, void* d_ws, size_t ws_size,
                              hipStream_t stream) {
  const float* nf         = (const float*)d_in[0];
  const float* nodes_mask = (const float*)d_in[3];
  const float* edges_mask = (const float*)d_in[4];
  const float* e_conv_w   = (const float*)d_in[5];
  const float* e_conv_b   = (const float*)d_in[6];
  const float* e_fc1_w    = (const float*)d_in[7];
  const float* e_fc1_b    = (const float*)d_in[8];
  const float* e_fc2_w    = (const float*)d_in[9];
  const float* e_fc2_b    = (const float*)d_in[10];
  const float* e_ln_g     = (const float*)d_in[11];
  const float* e_ln_b     = (const float*)d_in[12];
  const float* att_w      = (const float*)d_in[13];
  const float* att_b      = (const float*)d_in[14];
  const float* n_conv_w   = (const float*)d_in[15];
  const float* n_conv_b   = (const float*)d_in[16];
  const float* n_fc1_w    = (const float*)d_in[17];
  const float* n_fc1_b    = (const float*)d_in[18];
  const float* n_fc2_w    = (const float*)d_in[19];
  const float* n_fc2_b    = (const float*)d_in[20];
  const float* n_ln_g     = (const float*)d_in[21];
  const float* n_ln_b     = (const float*)d_in[22];
  const float* ne_conv_w  = (const float*)d_in[23];
  const float* ne_conv_b  = (const float*)d_in[24];
  const float* ne_fc1_w   = (const float*)d_in[25];
  const float* ne_fc1_b   = (const float*)d_in[26];
  const float* ne_fc2_w   = (const float*)d_in[27];
  const float* ne_fc2_b   = (const float*)d_in[28];
  const float* ne_ln_g    = (const float*)d_in[29];
  const float* ne_ln_b    = (const float*)d_in[30];
  char* ws = (char*)d_ws;
  float* out = (float*)d_out;

  prep_kernel<<<1024, 256, 0, stream>>>(nf, e_conv_w, e_fc1_w, e_fc2_w, n_conv_w,
                                        n_fc1_w, n_fc2_w, ne_conv_w, ne_fc1_w,
                                        ne_fc2_w, ws);
  edge_kernel<0><<<256, 256, 0, stream>>>(e_conv_b, e_fc1_b, e_fc2_b, edges_mask,
                                          e_ln_g, e_ln_b, att_w, att_b, out, ws);
  stats_kernel<<<1, 64, 0, stream>>>(ws);
  edge_kernel<1><<<256, 256, 0, stream>>>(e_conv_b, e_fc1_b, e_fc2_b, edges_mask,
                                          e_ln_g, e_ln_b, att_w, att_b, out, ws);
  node_kernel<<<2, 128, 0, stream>>>(nodes_mask, n_conv_b, n_fc1_b, n_fc2_b, n_ln_g,
                                     n_ln_b, ne_conv_b, ne_fc1_b, ne_fc2_b, ne_ln_g,
                                     ne_ln_b, out, ws);
}
